// UDecoder_26225070310021
// MI455X (gfx1250) — compile-verified, matrix operand order checked by reading, not run on hardware
//
#include <hip/hip_runtime.h>

// ---------------------------------------------------------------------------
// CDNA5 (gfx1250) transformer decoder block, bf16 WMMA everywhere.
// ---------------------------------------------------------------------------

typedef __attribute__((ext_vector_type(16))) __bf16 v16bf;
typedef __attribute__((ext_vector_type(8)))  __bf16 v8bf;
typedef __attribute__((ext_vector_type(8)))  float  v8f;
typedef int v4i_ __attribute__((vector_size(16)));   // matches builtin param type

#define BB   4
#define SS   2048
#define DD   512
#define HH   8
#define DKK  64
#define DFFN 2048
#define MM   (BB * SS)

#define WMMA_BF16(a, b, c) \
  __builtin_amdgcn_wmma_f32_16x16x32_bf16(false, (a), false, (b), (short)0, (c), false, false)

// Async global->LDS copy (CDNA5 GLOBAL_LOAD_ASYNC_TO_LDS_B128, ASYNCcnt).
// Signature (from hipcc diagnostic): (v4i addrspace(1)*, v4i addrspace(3)*,
// imm offset, imm cpol). Guarded: falls back to load+ds_store if absent.
#if __has_builtin(__builtin_amdgcn_global_load_async_to_lds_b128) && \
    __has_builtin(__builtin_amdgcn_s_wait_asynccnt)
#define HAVE_ASYNC_LDS 1
#else
#define HAVE_ASYNC_LDS 0
#endif

__device__ __forceinline__ void copy16B_g2l(const __bf16* g, __bf16* l) {
#if HAVE_ASYNC_LDS
  // Flat global address value == AS1 value; low 32 bits of a generic LDS
  // pointer == LDS byte offset == AS3 pointer value.
  __builtin_amdgcn_global_load_async_to_lds_b128(
      (__attribute__((address_space(1))) v4i_*)(unsigned long long)g,
      (__attribute__((address_space(3))) v4i_*)(unsigned int)(unsigned long long)l,
      0, 0);
#else
  *(v8bf*)l = *(const v8bf*)g;
#endif
}

__device__ __forceinline__ void staging_wait() {
#if HAVE_ASYNC_LDS
  __builtin_amdgcn_s_wait_asynccnt(0);
#endif
}

// Build a v16bf A/B fragment from two contiguous 8-element (16B) chunks:
// elements [0..7] from lo, [8..15] from lo+16 (ISA 16-bit operand layout:
// lanes 0-15 hold K=0..7 / 16..23; lanes 16-31 hold K=8..15 / 24..31 via the
// caller's half*8 base offset).
__device__ __forceinline__ v16bf make_frag(const __bf16* lo) {
  v8bf a = *(const v8bf*)lo;
  v8bf b = *(const v8bf*)(lo + 16);
  v16bf f;
#pragma unroll
  for (int i = 0; i < 8; ++i) { f[i] = a[i]; f[i + 8] = b[i]; }
  return f;
}

// ---------------------------------------------------------------------------
// Weight transpose + fp32 -> bf16 convert:  W[K][N]  ->  Wt[N][K]
// ---------------------------------------------------------------------------
__global__ __launch_bounds__(256) void wtrans_kernel(const float* __restrict__ W,
                                                     __bf16* __restrict__ Wt,
                                                     int K, int N) {
  int idx = blockIdx.x * 256 + threadIdx.x;
  if (idx >= K * N) return;
  int k = idx / N;
  int n = idx % N;
  Wt[(size_t)n * K + k] = (__bf16)W[idx];
}

// ---------------------------------------------------------------------------
// Row LayerNorm over D=512. One wave per row (16 floats/lane via 4x float4).
// ---------------------------------------------------------------------------
__global__ __launch_bounds__(256) void ln512_kernel(const float* __restrict__ in,
                                                    const float* __restrict__ g,
                                                    const float* __restrict__ be,
                                                    float* __restrict__ outF,
                                                    __bf16* __restrict__ outB,
                                                    int rows) {
  const int wave = threadIdx.x >> 5;
  const int lane = threadIdx.x & 31;
  const int row  = blockIdx.x * 8 + wave;
  if (row >= rows) return;

  const float4* p = (const float4*)(in + (size_t)row * DD);
  float4 c[4];
  float s = 0.f;
#pragma unroll
  for (int i = 0; i < 4; ++i) {
    c[i] = p[lane + i * 32];
    s += c[i].x + c[i].y + c[i].z + c[i].w;
  }
#pragma unroll
  for (int sft = 16; sft >= 1; sft >>= 1) s += __shfl_xor(s, sft, 32);
  const float mean = s * (1.f / DD);

  float vs = 0.f;
#pragma unroll
  for (int i = 0; i < 4; ++i) {
    float dx = c[i].x - mean, dy = c[i].y - mean;
    float dz = c[i].z - mean, dw = c[i].w - mean;
    vs += dx * dx + dy * dy + dz * dz + dw * dw;
  }
#pragma unroll
  for (int sft = 16; sft >= 1; sft >>= 1) vs += __shfl_xor(vs, sft, 32);
  const float inv = rsqrtf(vs * (1.f / DD) + 1e-5f);

#pragma unroll
  for (int i = 0; i < 4; ++i) {
    const int base = 4 * (lane + i * 32);
    float vv[4] = {c[i].x, c[i].y, c[i].z, c[i].w};
#pragma unroll
    for (int jj = 0; jj < 4; ++jj) {
      const int col = base + jj;
      const float outv = g[col] * (vv[jj] - mean) * inv + be[col];
      const size_t idx = (size_t)row * DD + col;
      if (outF) outF[idx] = outv;
      if (outB) outB[idx] = (__bf16)outv;
    }
  }
}

// ---------------------------------------------------------------------------
// Tiled bf16 GEMM: out[M,N] = A[M,K](bf16) @ Wt[N,K](bf16)^T + bias (+resid)
// Block: 128 threads (4 waves), block tile 128(M) x 64(N), BK = 32.
// Per wave: 32(M) x 64(N) -> 2 A-frags x 4 B-frags = 8 WMMAs per K-step
// (1.5 LDS fragment loads per WMMA). Staging via async global->LDS when
// available.
// ---------------------------------------------------------------------------
#define AS_STRIDE 40   // padded LDS row stride (elems), 80B -> 16B aligned
__global__ __launch_bounds__(128) void gemm_bf16_kernel(
    const __bf16* __restrict__ A, const __bf16* __restrict__ Wt,
    const float* __restrict__ bias, const float* __restrict__ resid,
    float* __restrict__ outF, __bf16* __restrict__ outB,
    int M, int N, int K, int doRelu) {
  const int tid  = threadIdx.x;
  const int wave = tid >> 5;
  const int lane = tid & 31;
  const int hf   = lane >> 4;
  const int l16  = lane & 15;
  const int bm   = blockIdx.x * 128;
  const int bn   = blockIdx.y * 64;

  __shared__ __align__(16) __bf16 As[128 * AS_STRIDE];
  __shared__ __align__(16) __bf16 Bs[64 * AS_STRIDE];

  v8f acc[2][4];
#pragma unroll
  for (int ms = 0; ms < 2; ++ms)
#pragma unroll
    for (int t = 0; t < 4; ++t)
#pragma unroll
      for (int i = 0; i < 8; ++i) acc[ms][t][i] = 0.f;

  const int m0 = bm + wave * 32;

  for (int k0 = 0; k0 < K; k0 += 32) {
    {
      // A tile: 128 rows x 32 cols; thread t stages its whole row (4 x 16B).
      const __bf16* srcA = A + (size_t)(bm + tid) * K + k0;
      __bf16* dstA = &As[tid * AS_STRIDE];
#pragma unroll
      for (int c = 0; c < 4; ++c) copy16B_g2l(srcA + c * 8, dstA + c * 8);
      // B tile: 64 rows x 32 cols; thread t stages half a row (2 x 16B).
      const int rn = tid >> 1;
      const int cn = (tid & 1) * 16;
      const __bf16* srcB = Wt + (size_t)(bn + rn) * K + k0 + cn;
      __bf16* dstB = &Bs[rn * AS_STRIDE + cn];
      copy16B_g2l(srcB,     dstB);
      copy16B_g2l(srcB + 8, dstB + 8);
      if (k0 + 32 < K) __builtin_prefetch(srcA + 32, 0, 1);  // global_prefetch_b8
    }
    staging_wait();
    __syncthreads();

    v16bf afrag[2];
#pragma unroll
    for (int ms = 0; ms < 2; ++ms)
      afrag[ms] = make_frag(&As[(wave * 32 + ms * 16 + l16) * AS_STRIDE + hf * 8]);
#pragma unroll
    for (int t = 0; t < 4; ++t) {
      v16bf bfrag = make_frag(&Bs[(t * 16 + l16) * AS_STRIDE + hf * 8]);
#pragma unroll
      for (int ms = 0; ms < 2; ++ms)
        acc[ms][t] = WMMA_BF16(afrag[ms], bfrag, acc[ms][t]);
    }
    __syncthreads();
  }

  // Epilogue. C layout: VGPR r, lanes 0-15 -> M=r, lanes 16-31 -> M=8+r; N=l16.
#pragma unroll
  for (int ms = 0; ms < 2; ++ms) {
#pragma unroll
    for (int t = 0; t < 4; ++t) {
      const int col = bn + t * 16 + l16;
      const float bb = bias[col];
#pragma unroll
      for (int r = 0; r < 8; ++r) {
        const int row = m0 + ms * 16 + r + hf * 8;
        float v = acc[ms][t][r] + bb;
        if (doRelu) v = fmaxf(v, 0.f);
        const size_t idx = (size_t)row * N + col;
        if (resid) v += resid[idx];
        if (outF) outF[idx] = v;
        if (outB) outB[idx] = (__bf16)v;
      }
    }
  }
}

// ---------------------------------------------------------------------------
// Causal flash attention, bf16 WMMA. Block = 4 waves, 64 query rows shared.
// Per 32-key step: stage K (async, row-major) + V^T in LDS; Q.K^T = 4 WMMAs;
// online softmax (stats per (r, lane-half) row == C-matrix row mapping,
// shfl_xor over 16 lanes); P round-trips LDS (C-layout -> A-layout);
// P.V = 4 WMMAs. q/k/v/o layout: [b*S + s][h*DK + dk] (stride D).
// ---------------------------------------------------------------------------
#define KS_STRIDE 72
#define VT_STRIDE 40
#define PW_STRIDE 40
__global__ __launch_bounds__(128) void attn_kernel(const __bf16* __restrict__ qg,
                                                   const __bf16* __restrict__ kg,
                                                   const __bf16* __restrict__ vg,
                                                   __bf16* __restrict__ og) {
  const int wave = threadIdx.x >> 5;
  const int lane = threadIdx.x & 31;
  const int hf   = lane >> 4;
  const int l16  = lane & 15;
  const int b    = blockIdx.z;
  const int h    = blockIdx.y;
  const int q0   = blockIdx.x * 64;        // block's query supertile
  const int qw   = q0 + wave * 16;         // this wave's 16 query rows
  const size_t seqBase = (size_t)b * SS * DD + (size_t)h * DKK;

  __shared__ __align__(16) __bf16 Ks[32 * KS_STRIDE];    // [key][dk]
  __shared__ __align__(16) __bf16 Vt[64 * VT_STRIDE];    // [dk][key]
  __shared__ __align__(16) __bf16 Pw[4][16 * PW_STRIDE]; // per-wave P

  // Q fragments (registers for whole kernel): 16 rows x 64 dk.
  v16bf aQ[2];
#pragma unroll
  for (int ks = 0; ks < 2; ++ks) {
    const __bf16* p = qg + seqBase + (size_t)(qw + l16) * DD + ks * 32 + hf * 8;
    aQ[ks] = make_frag(p);
  }

  float rm[8], rl[8];
  v8f accO[4];
#pragma unroll
  for (int r = 0; r < 8; ++r) { rm[r] = -__builtin_inff(); rl[r] = 0.f; }
#pragma unroll
  for (int t = 0; t < 4; ++t)
#pragma unroll
    for (int i = 0; i < 8; ++i) accO[t][i] = 0.f;

  const int jmax = (q0 + 63) >> 5;   // shared across waves -> uniform barriers
  const float scale = 0.125f;        // 1/sqrt(64)

  for (int j = 0; j <= jmax; ++j) {
    const int kb = j * 32;
    {
      // 32 key rows; thread handles (key r, dk chunk c..c+15).
      const int r = threadIdx.x >> 2;
      const int c = (threadIdx.x & 3) * 16;
      const __bf16* ksrc = kg + seqBase + (size_t)(kb + r) * DD + c;
      copy16B_g2l(ksrc,     &Ks[r * KS_STRIDE + c]);
      copy16B_g2l(ksrc + 8, &Ks[r * KS_STRIDE + c + 8]);
      // V^T staging (elementwise transpose -> plain VALU path).
      const __bf16* vsrc = vg + seqBase + (size_t)(kb + r) * DD + c;
      v8bf va = *(const v8bf*)vsrc;
      v8bf vb = *(const v8bf*)(vsrc + 8);
#pragma unroll
      for (int i = 0; i < 8; ++i) {
        Vt[(c + i) * VT_STRIDE + r]     = va[i];
        Vt[(c + 8 + i) * VT_STRIDE + r] = vb[i];
      }
    }
    staging_wait();
    __syncthreads();

    // Scores: two 16x16 sub-tiles over 32 keys, each 2 K-steps over dk=64.
    v8f sc[2];
#pragma unroll
    for (int sub = 0; sub < 2; ++sub) {
#pragma unroll
      for (int i = 0; i < 8; ++i) sc[sub][i] = 0.f;
#pragma unroll
      for (int ks = 0; ks < 2; ++ks) {
        v16bf bK = make_frag(&Ks[(sub * 16 + l16) * KS_STRIDE + ks * 32 + hf * 8]);
        sc[sub] = WMMA_BF16(aQ[ks], bK, sc[sub]);
      }
    }

    // Online softmax update per row (r, half).
#pragma unroll
    for (int r = 0; r < 8; ++r) {
      const int qrow = qw + r + hf * 8;
      float v0 = sc[0][r] * scale;
      float v1 = sc[1][r] * scale;
      if (kb + l16 > qrow)      v0 = -__builtin_inff();
      if (kb + 16 + l16 > qrow) v1 = -__builtin_inff();
      float m = fmaxf(v0, v1);
#pragma unroll
      for (int sft = 8; sft >= 1; sft >>= 1) m = fmaxf(m, __shfl_xor(m, sft, 32));
      const float nm = fmaxf(rm[r], m);
      const float p0 = __expf(v0 - nm);
      const float p1 = __expf(v1 - nm);
      float sum = p0 + p1;
#pragma unroll
      for (int sft = 8; sft >= 1; sft >>= 1) sum += __shfl_xor(sum, sft, 32);
      const float f = __expf(rm[r] - nm);   // exp(-inf - finite) = 0 at start
      rm[r] = nm;
      rl[r] = rl[r] * f + sum;
#pragma unroll
      for (int t = 0; t < 4; ++t) accO[t][r] *= f;
      const int prow = r + hf * 8;
      Pw[wave][prow * PW_STRIDE + l16]      = (__bf16)p0;
      Pw[wave][prow * PW_STRIDE + 16 + l16] = (__bf16)p1;
    }

    // P @ V (same-wave LDS RAW: per-wave DS pipeline is in-order; load-result
    // DScnt dependencies are compiler-managed).
    {
      v16bf aP = make_frag(&Pw[wave][l16 * PW_STRIDE + hf * 8]);
#pragma unroll
      for (int t = 0; t < 4; ++t) {
        v16bf bV = make_frag(&Vt[(t * 16 + l16) * VT_STRIDE + hf * 8]);
        accO[t] = WMMA_BF16(aP, bV, accO[t]);
      }
    }
    __syncthreads();   // before next iteration overwrites Ks/Vt
  }

  // Normalize and store bf16 output.
#pragma unroll
  for (int t = 0; t < 4; ++t) {
#pragma unroll
    for (int r = 0; r < 8; ++r) {
      const int row = qw + r + hf * 8;
      const float inv = 1.0f / rl[r];
      og[seqBase + (size_t)row * DD + t * 16 + l16] = (__bf16)(accO[t][r] * inv);
    }
  }
}

// ---------------------------------------------------------------------------
// Launcher
// ---------------------------------------------------------------------------
extern "C" void kernel_launch(void* const* d_in, const int* in_sizes, int n_in,
                              void* d_out, int out_size, void* d_ws, size_t ws_size,
                              hipStream_t stream) {
  (void)in_sizes; (void)n_in; (void)out_size; (void)ws_size;

  const float* x    = (const float*)d_in[0];
  const float* Wq   = (const float*)d_in[1];
  const float* bq   = (const float*)d_in[2];
  const float* Wk   = (const float*)d_in[3];
  const float* bk   = (const float*)d_in[4];
  const float* Wv   = (const float*)d_in[5];
  const float* bv   = (const float*)d_in[6];
  const float* Wo   = (const float*)d_in[7];
  const float* bo   = (const float*)d_in[8];
  const float* W1   = (const float*)d_in[9];
  const float* b1   = (const float*)d_in[10];
  const float* W2   = (const float*)d_in[11];
  const float* b2   = (const float*)d_in[12];
  const float* ln0g = (const float*)d_in[13];
  const float* ln0b = (const float*)d_in[14];
  const float* ln2g = (const float*)d_in[15];
  const float* ln2b = (const float*)d_in[16];
  const float* lnfg = (const float*)d_in[17];
  const float* lnfb = (const float*)d_in[18];

  char* ws = (char*)d_ws;
  size_t off = 0;
  auto take = [&](size_t bytes) -> void* {
    void* p = ws + off;
    off += (bytes + 255) & ~(size_t)255;
    return p;
  };

  __bf16* xln = (__bf16*)take((size_t)MM * DD * 2);
  __bf16* wqT = (__bf16*)take((size_t)DD * DD * 2);
  __bf16* wkT = (__bf16*)take((size_t)DD * DD * 2);
  __bf16* wvT = (__bf16*)take((size_t)DD * DD * 2);
  __bf16* woT = (__bf16*)take((size_t)DD * DD * 2);
  __bf16* w1T = (__bf16*)take((size_t)DD * DFFN * 2);
  __bf16* w2T = (__bf16*)take((size_t)DFFN * DD * 2);
  __bf16* qb  = (__bf16*)take((size_t)MM * DD * 2);
  __bf16* kb2 = (__bf16*)take((size_t)MM * DD * 2);
  __bf16* vb2 = (__bf16*)take((size_t)MM * DD * 2);
  __bf16* wvb = (__bf16*)take((size_t)MM * DD * 2);
  float*  x1  = (float*) take((size_t)MM * DD * 4);
  __bf16* yb  = (__bf16*)take((size_t)MM * DD * 2);
  __bf16* hb  = (__bf16*)take((size_t)MM * DFFN * 2);
  float*  x2  = (float*) take((size_t)MM * DD * 4);

  // Weight transpose/convert to bf16 [N][K].
  wtrans_kernel<<<(DD * DD + 255) / 256, 256, 0, stream>>>(Wq, wqT, DD, DD);
  wtrans_kernel<<<(DD * DD + 255) / 256, 256, 0, stream>>>(Wk, wkT, DD, DD);
  wtrans_kernel<<<(DD * DD + 255) / 256, 256, 0, stream>>>(Wv, wvT, DD, DD);
  wtrans_kernel<<<(DD * DD + 255) / 256, 256, 0, stream>>>(Wo, woT, DD, DD);
  wtrans_kernel<<<(DD * DFFN + 255) / 256, 256, 0, stream>>>(W1, w1T, DD, DFFN);
  wtrans_kernel<<<(DFFN * DD + 255) / 256, 256, 0, stream>>>(W2, w2T, DFFN, DD);

  // LN0 -> bf16 activations.
  ln512_kernel<<<MM / 8, 256, 0, stream>>>(x, ln0g, ln0b, nullptr, xln, MM);

  // Q/K/V projections.
  dim3 gD(MM / 128, DD / 64);
  gemm_bf16_kernel<<<gD, 128, 0, stream>>>(xln, wqT, bq, nullptr, nullptr, qb,  MM, DD, DD, 0);
  gemm_bf16_kernel<<<gD, 128, 0, stream>>>(xln, wkT, bk, nullptr, nullptr, kb2, MM, DD, DD, 0);
  gemm_bf16_kernel<<<gD, 128, 0, stream>>>(xln, wvT, bv, nullptr, nullptr, vb2, MM, DD, DD, 0);

  // Causal flash attention.
  dim3 gA(SS / 64, HH, BB);
  attn_kernel<<<gA, 128, 0, stream>>>(qb, kb2, vb2, wvb);

  // O projection + residual -> x1 (fp32).
  gemm_bf16_kernel<<<gD, 128, 0, stream>>>(wvb, woT, bo, x, x1, nullptr, MM, DD, DD, 0);

  // LN2 -> bf16.
  ln512_kernel<<<MM / 8, 256, 0, stream>>>(x1, ln2g, ln2b, nullptr, yb, MM);

  // FFN.
  dim3 gF(MM / 128, DFFN / 64);
  gemm_bf16_kernel<<<gF, 128, 0, stream>>>(yb, w1T, b1, nullptr, nullptr, hb, MM, DFFN, DD, 1);
  gemm_bf16_kernel<<<gD, 128, 0, stream>>>(hb, w2T, b2, x1, x2, nullptr, MM, DD, DFFN, 0);

  // Final LN -> fp32 output.
  ln512_kernel<<<MM / 8, 256, 0, stream>>>(x2, lnfg, lnfb, (float*)d_out, nullptr, MM);
}
